// HybridDecompExtractor_84817014162116
// MI455X (gfx1250) — matrix operand order repaired and checked by reading, not compiled, not run on hardware
//
#include <hip/hip_runtime.h>
#include <hip/hip_bf16.h>

typedef __attribute__((ext_vector_type(16))) _Float16 v16h;
typedef __attribute__((ext_vector_type(8)))  _Float16 v8h;
typedef __attribute__((ext_vector_type(8)))  float    v8f;
typedef __attribute__((ext_vector_type(4)))  int      v4i;

#define B_DIM 8192
#define L_DIM 4096
#define P_DIM 720
#define H_DIM 2048
#define K_TOT 8192   // L + 2H
#define INV_SQRT2f 0.70710678118654752440f

#if __has_builtin(__builtin_amdgcn_global_load_async_to_lds_b128)
#define USE_ASYNC_LDS 1
#else
#define USE_ASYNC_LDS 0
#endif

__device__ __forceinline__ void wait_async0() {
#if __has_builtin(__builtin_amdgcn_s_wait_asynccnt)
    __builtin_amdgcn_s_wait_asynccnt(0);
#else
    asm volatile("s_wait_asynccnt 0x0" ::: "memory");
#endif
}

#if USE_ASYNC_LDS
// 16-byte async global->LDS copy (CDNA5, tracked by ASYNCcnt).
// Signature discovered from hipcc diagnostics: param0 is a v4i pointer
// (global source), param1 the LDS destination, then imm offset + imm cpol.
__device__ __forceinline__ void async_copy16(const void* g, void* l) {
    __builtin_amdgcn_global_load_async_to_lds_b128(
        (__attribute__((address_space(1))) v4i*)g,
        (__attribute__((address_space(3))) v4i*)l,
        0, 0);
}
#endif

// ---------------- Kernel 1: fuse + convert weights to f16 -----------------
__global__ __launch_bounds__(256) void prep_weights_kernel(
    const float* __restrict__ Wt, const float* __restrict__ Wl,
    const float* __restrict__ Wh, _Float16* __restrict__ Wf)
{
    size_t idx = (size_t)blockIdx.x * 256 + threadIdx.x;
    if (idx >= (size_t)P_DIM * K_TOT) return;
    int p = (int)(idx >> 13);
    int k = (int)(idx & (K_TOT - 1));
    float v;
    if (k < L_DIM)            v = Wt[(size_t)p * L_DIM + k];
    else if (k < L_DIM+H_DIM) v = Wl[(size_t)p * H_DIM + (k - L_DIM)];
    else                      v = Wh[(size_t)p * H_DIM + (k - L_DIM - H_DIM)];
    Wf[idx] = (_Float16)v;
}

// ---------------- Kernel 2: per-row decomposition -------------------------
__global__ __launch_bounds__(256) void prep_rows_kernel(
    const float* __restrict__ x, _Float16* __restrict__ A,
    float* __restrict__ stdbuf)
{
    __shared__ float sx[L_DIM];
    __shared__ float red[256];
    __shared__ float red2[256];
    __shared__ float s_mean, s_std;

    const int b = blockIdx.x;
    const int t = threadIdx.x;
    const float4* x4 = (const float4*)(x + (size_t)b * L_DIM);
    for (int i = t; i < L_DIM / 4; i += 256) ((float4*)sx)[i] = x4[i];
    __syncthreads();

    float tr[16], rs[16];
    float sum = 0.f, sum2 = 0.f;
    const int base = t * 16;
    #pragma unroll
    for (int u = 0; u < 16; ++u) {
        int i = base + u;
        float acc = 0.f;
        #pragma unroll
        for (int j = -12; j <= 12; ++j) {
            int idx = i + j;
            idx = idx < 0 ? 0 : (idx > L_DIM - 1 ? L_DIM - 1 : idx);
            acc += sx[idx];
        }
        float trd = acc * (1.0f / 25.0f);
        tr[u] = trd;
        float r = sx[i] - trd;
        rs[u] = r;
        sum += r; sum2 += r * r;
    }
    red[t] = sum; red2[t] = sum2;
    __syncthreads();
    for (int s = 128; s > 0; s >>= 1) {
        if (t < s) { red[t] += red[t + s]; red2[t] += red2[t + s]; }
        __syncthreads();
    }
    if (t == 0) {
        float mean = red[0] / (float)L_DIM;
        float var  = (red2[0] - (float)L_DIM * mean * mean) / (float)(L_DIM - 1);
        var = var < 0.f ? 0.f : var;
        float sd = sqrtf(var) + 1e-5f;
        s_mean = mean; s_std = sd;
        stdbuf[b] = sd;
    }
    __syncthreads();
    const float mean = s_mean;

    _Float16* Ar = A + (size_t)b * K_TOT;
    #pragma unroll
    for (int u = 0; u < 16; ++u) Ar[base + u] = (_Float16)tr[u];
    #pragma unroll
    for (int j = 0; j < 8; ++j) {
        float e = rs[2 * j], o = rs[2 * j + 1];
        Ar[L_DIM + t * 8 + j]         = (_Float16)((e + o - 2.f * mean) * INV_SQRT2f);
        Ar[L_DIM + H_DIM + t * 8 + j] = (_Float16)((e - o) * INV_SQRT2f);
    }
}

// ---------------- Kernel 3: fused WMMA GEMM + bias epilogue ---------------
// C[b,p] = sum_k A[b,k]*Wcat[p,k] + b_trend[p] + (b_lo[p]+b_hi[p])*std[b]
#define BM 128
#define BN 64
#define KB 64
#define LDS_STRIDE 72   // 64 + 8 halfs padding (144B rows, 16B aligned)

__global__ __launch_bounds__(256) void gemm_kernel(
    const _Float16* __restrict__ A, const _Float16* __restrict__ W,
    const float* __restrict__ bt, const float* __restrict__ bl,
    const float* __restrict__ bh, const float* __restrict__ stdbuf,
    float* __restrict__ out)
{
    __shared__ __align__(16) _Float16 As[2][BM][LDS_STRIDE];
    __shared__ __align__(16) _Float16 Ws[2][BN][LDS_STRIDE];

    const int t      = threadIdx.x;
    const int mblock = blockIdx.x * BM;
    const int nblock = blockIdx.y * BN;
    const int lane   = t & 31;
    const int w      = t >> 5;
    const int wm     = (w & 3) * 32;   // wave M offset inside tile (4 waves)
    const int wn     = (w >> 2) * 32;  // wave N offset inside tile (2 waves)

    v8f c00 = {}, c01 = {}, c10 = {}, c11 = {};

    const int NT = K_TOT / KB;  // 128 k-tiles

    // Zero the N-tail rows of both W stages once (async copies can't zero).
    {
        #pragma unroll
        for (int i = 0; i < 2; ++i) {
            int c = t + 256 * i; int row = c >> 3; int kk = (c & 7) * 8;
            if (nblock + row >= P_DIM) {
                *(uint4*)&Ws[0][row][kk] = make_uint4(0u, 0u, 0u, 0u);
                *(uint4*)&Ws[1][row][kk] = make_uint4(0u, 0u, 0u, 0u);
            }
        }
    }

#if USE_ASYNC_LDS
    // Issue CDNA5 async global->LDS copies for k-tile kt into stage s.
    auto issueTile = [&](int kt, int s) {
        #pragma unroll
        for (int i = 0; i < 4; ++i) {
            int c = t + 256 * i; int row = c >> 3; int kk = (c & 7) * 8;
            const _Float16* g = A + (size_t)(mblock + row) * K_TOT + kt * KB + kk;
            async_copy16(g, &As[s][row][kk]);
        }
        #pragma unroll
        for (int i = 0; i < 2; ++i) {
            int c = t + 256 * i; int row = c >> 3; int kk = (c & 7) * 8;
            int pidx = nblock + row;
            if (pidx < P_DIM) {
                const _Float16* g = W + (size_t)pidx * K_TOT + kt * KB + kk;
                async_copy16(g, &Ws[s][row][kk]);
            }
        }
    };
#else
    // Fallback: synchronous register staging (no prefetch to limit pressure).
    auto issueTile = [&](int kt, int s) {
        #pragma unroll
        for (int i = 0; i < 4; ++i) {
            int c = t + 256 * i; int row = c >> 3; int kk = (c & 7) * 8;
            uint4 v = *(const uint4*)(A + (size_t)(mblock + row) * K_TOT + kt * KB + kk);
            *(uint4*)&As[s][row][kk] = v;
        }
        #pragma unroll
        for (int i = 0; i < 2; ++i) {
            int c = t + 256 * i; int row = c >> 3; int kk = (c & 7) * 8;
            int pidx = nblock + row;
            if (pidx < P_DIM) {
                uint4 v = *(const uint4*)(W + (size_t)pidx * K_TOT + kt * KB + kk);
                *(uint4*)&Ws[s][row][kk] = v;
            }
        }
    };
#endif

    issueTile(0, 0);
#if USE_ASYNC_LDS
    wait_async0();
#endif
    __syncthreads();

    const int am  = lane & 15;          // A fragment: lane -> M row
    const int akb = (lane >> 4) * 8;    // K base 0/8 (second half at +16)
    const int bn  = lane & 15;          // B fragment: lane -> N col
    const int bkb = (lane >> 4) * 8;    // K base 0/8 (second half at +16)

    for (int kt = 0; kt < NT; ++kt) {
        const int s = kt & 1;
        // Buffer s^1 was last read in iteration kt-1 (fenced by its barrier),
        // so it is safe to start overwriting it now.
        if (kt + 1 < NT) issueTile(kt + 1, s ^ 1);

        #pragma unroll
        for (int ks = 0; ks < 2; ++ks) {
            const int k0 = ks * 32;
            // ---- A fragments (16x32 f16, documented lane/K layout) ----
            v8h a0l = *(const v8h*)&As[s][wm + am     ][k0 + akb];
            v8h a0h = *(const v8h*)&As[s][wm + am     ][k0 + akb + 16];
            v8h a1l = *(const v8h*)&As[s][wm + 16 + am][k0 + akb];
            v8h a1h = *(const v8h*)&As[s][wm + 16 + am][k0 + akb + 16];
            v16h a0 = __builtin_shufflevector(a0l, a0h, 0,1,2,3,4,5,6,7,8,9,10,11,12,13,14,15);
            v16h a1 = __builtin_shufflevector(a1l, a1h, 0,1,2,3,4,5,6,7,8,9,10,11,12,13,14,15);
            // ---- B fragments (32x16 f16, lane=N, same K layout as A) ----
            v8h b0l = *(const v8h*)&Ws[s][wn + bn     ][k0 + bkb];
            v8h b0h = *(const v8h*)&Ws[s][wn + bn     ][k0 + bkb + 16];
            v8h b1l = *(const v8h*)&Ws[s][wn + 16 + bn][k0 + bkb];
            v8h b1h = *(const v8h*)&Ws[s][wn + 16 + bn][k0 + bkb + 16];
            v16h b0 = __builtin_shufflevector(b0l, b0h, 0,1,2,3,4,5,6,7,8,9,10,11,12,13,14,15);
            v16h b1 = __builtin_shufflevector(b1l, b1h, 0,1,2,3,4,5,6,7,8,9,10,11,12,13,14,15);

            c00 = __builtin_amdgcn_wmma_f32_16x16x32_f16(false, a0, false, b0, (short)0, c00, false, false);
            c01 = __builtin_amdgcn_wmma_f32_16x16x32_f16(false, a0, false, b1, (short)0, c01, false, false);
            c10 = __builtin_amdgcn_wmma_f32_16x16x32_f16(false, a1, false, b0, (short)0, c10, false, false);
            c11 = __builtin_amdgcn_wmma_f32_16x16x32_f16(false, a1, false, b1, (short)0, c11, false, false);
        }

        if (kt + 1 < NT) {
#if USE_ASYNC_LDS
            wait_async0();       // next stage fully landed in LDS
            __syncthreads();     // all waves done reading stage s
#else
            __syncthreads();     // writers for kt+1 finished before readers move on
            __syncthreads();
#endif
        }
    }

    // ---- Epilogue: C layout VGPR r -> M = r + 8*(lane>=16), N = lane&15 ----
    const int half = lane >> 4;
    const int nloc = lane & 15;
    auto storeFrag = [&](const v8f& cf, int fm, int fn) {
        int col = nblock + wn + fn * 16 + nloc;
        if (col >= P_DIM) return;
        float bias_t = bt[col];
        float bias_r = bl[col] + bh[col];
        #pragma unroll
        for (int r = 0; r < 8; ++r) {
            int row = mblock + wm + fm * 16 + r + 8 * half;
            float sd = stdbuf[row];
            out[(size_t)row * P_DIM + col] = cf[r] + bias_t + bias_r * sd;
        }
    };
    storeFrag(c00, 0, 0);
    storeFrag(c01, 0, 1);
    storeFrag(c10, 1, 0);
    storeFrag(c11, 1, 1);
}

// ---------------------------------------------------------------------------
extern "C" void kernel_launch(void* const* d_in, const int* in_sizes, int n_in,
                              void* d_out, int out_size, void* d_ws, size_t ws_size,
                              hipStream_t stream) {
    const float* x  = (const float*)d_in[0];
    const float* Wt = (const float*)d_in[1];
    const float* bt = (const float*)d_in[2];
    const float* Wl = (const float*)d_in[3];
    const float* bl = (const float*)d_in[4];
    const float* Wh = (const float*)d_in[5];
    const float* bh = (const float*)d_in[6];
    float* out = (float*)d_out;

    char* ws = (char*)d_ws;
    _Float16* Af = (_Float16*)ws;                                      // 8192*8192*2 B
    _Float16* Wf = (_Float16*)(ws + (size_t)B_DIM * K_TOT * 2);        // 720*8192*2 B
    float* stdbuf = (float*)(ws + (size_t)B_DIM * K_TOT * 2
                                + (size_t)P_DIM * K_TOT * 2);          // 8192*4 B

    {
        size_t total = (size_t)P_DIM * K_TOT;
        int blocks = (int)((total + 255) / 256);
        prep_weights_kernel<<<blocks, 256, 0, stream>>>(Wt, Wl, Wh, Wf);
    }
    prep_rows_kernel<<<B_DIM, 256, 0, stream>>>(x, Af, stdbuf);
    {
        dim3 grid(B_DIM / BM, (P_DIM + BN - 1) / BN);   // 64 x 12
        gemm_kernel<<<grid, 256, 0, stream>>>(Af, Wf, bt, bl, bh, stdbuf, out);
    }
}
